// Block_65180423685211
// MI455X (gfx1250) — matrix operand order repaired and checked by reading, not compiled, not run on hardware
//
#include <hip/hip_runtime.h>
#include <math.h>

// ---------------- constants ----------------
#define Bn 8
#define Cc 256
#define Hh 64
#define Ww 64
#define Tt (Bn*Hh*Ww)          // 32768 tokens
#define NHEAD 8
#define DHEAD 32
#define NWIN 8
#define NREG 64                // regions per image
#define RSQ 64                 // tokens per region
#define TOPK 4
#define KSEL 32                // top-k inside attention = TOPK*RSQ/8
#define HID 768

typedef __attribute__((ext_vector_type(16))) __bf16 v16bf;
typedef __attribute__((ext_vector_type(8)))  float  v8f;
typedef __attribute__((ext_vector_type(8)))  unsigned short us8;

// ---------------- helpers ----------------
__device__ __forceinline__ unsigned short f2bf(float f) {
    unsigned int u = __float_as_uint(f);
    u += 0x7FFFu + ((u >> 16) & 1u);          // RNE
    return (unsigned short)(u >> 16);
}
__device__ __forceinline__ float bf2f(unsigned short h) {
    return __uint_as_float(((unsigned int)h) << 16);
}
// wave-relative LDS byte offset of a __shared__ object (flat addr low 32 bits)
__device__ __forceinline__ unsigned lds_off(const void* p) {
    return (unsigned)(size_t)p;
}
// CDNA5 async memory->LDS copies (ASYNCcnt tracked), saddr addressing mode:
// mem_addr = SADDR + VGPR32 ; LDS addr = LDS_BASE + VDST_VGPR
__device__ __forceinline__ void async_ld_b128(unsigned lds, unsigned voff, const void* base) {
    asm volatile("global_load_async_to_lds_b128 %0, %1, %2"
                 :: "v"(lds), "v"(voff), "s"(base) : "memory");
}
__device__ __forceinline__ void async_ld_b64(unsigned lds, unsigned voff, const void* base) {
    asm volatile("global_load_async_to_lds_b64 %0, %1, %2"
                 :: "v"(lds), "v"(voff), "s"(base) : "memory");
}
__device__ __forceinline__ void wait_async0() {
    asm volatile("s_wait_asynccnt 0x0" ::: "memory");
}

// A fragment: 16x32 bf16 (M x K), ISA 7.12.2 16-bit A layout.
// lane<16: M=lane, K in {0..7,16..23}; lane>=16: same M, K in {8..15,24..31}
__device__ __forceinline__ v16bf load_a_frag(const unsigned short* S, int rowBase, int lane) {
    int h = lane >> 4, r = lane & 15;
    const unsigned short* rp = S + (rowBase + r) * 32;
    union { unsigned int w[8]; v16bf v; } u;
#pragma unroll
    for (int vg = 0; vg < 8; ++vg) {
        int kk = (vg < 4) ? (h * 8 + vg * 2) : (16 + h * 8 + (vg - 4) * 2);
        u.w[vg] = *(const unsigned int*)(rp + kk);
    }
    return u.v;
}
// B fragment: 32x16 bf16 (K x N); S holds tile rows as [n][k] (k-stride 32).
// lane<16: N=lane, K in {0..7,16..23}; lane>=16: N=lane-16, K in {8..15,24..31}
__device__ __forceinline__ v16bf load_b_frag(const unsigned short* S, int colBase, int lane) {
    int h = lane >> 4, c = lane & 15;
    const unsigned short* rp = S + (colBase + c) * 32;
    union { unsigned int w[8]; v16bf v; } u;
#pragma unroll
    for (int i = 0; i < 8; ++i) u.w[i] = *(const unsigned int*)(rp + ((i < 4) ? (h * 8 + i * 2) : (16 + h * 8 + (i - 4) * 2)));
    return u.v;
}
__device__ __forceinline__ v8f wmma_bf16(v16bf a, v16bf b, v8f c) {
    return __builtin_amdgcn_wmma_f32_16x16x32_bf16(false, a, false, b, (short)0, c, false, false);
}
// stage one 128x32 bf16 A tile + 128x32 bf16 B tile via async-to-LDS
__device__ __forceinline__ void stage_tiles(
        const unsigned short* A, const unsigned short* W, int K,
        int m0, int n0, int k0, unsigned asBase, unsigned bsBase, int tid) {
    int e = tid * 8;
#pragma unroll
    for (int rep = 0; rep < 2; ++rep) {
        int row = e >> 5, col = e & 31;
        unsigned gA = (unsigned)(((m0 + row) * K + k0 + col) * 2);
        unsigned gB = (unsigned)(((n0 + row) * K + k0 + col) * 2);
        async_ld_b128(asBase + (unsigned)e * 2u, gA, A);
        async_ld_b128(bsBase + (unsigned)e * 2u, gB, W);
        e += 2048;
    }
}

// ---------------- weight conversion ----------------
__global__ void cvt_bf16(const float* __restrict__ in, unsigned short* __restrict__ out, int n) {
    int i = blockIdx.x * 256 + threadIdx.x;
    if (i < n) out[i] = f2bf(in[i]);
}
// in: [R, Ccols] row-major ; out: [Ccols, R]
__global__ void cvt_bf16_T(const float* __restrict__ in, unsigned short* __restrict__ out, int R, int Ccols) {
    int i = blockIdx.x * 256 + threadIdx.x;
    if (i < R * Ccols) {
        int r = i / Ccols, c = i - r * Ccols;
        out[(size_t)c * R + r] = f2bf(in[i]);
    }
}

// ---------------- pos dwconv 3x3 (NCHW in) -> NHWC residual stream ----------------
__global__ __launch_bounds__(256) void posconv_kernel(
        const float* __restrict__ x, const float* __restrict__ w,
        const float* __restrict__ bias, float* __restrict__ x1) {
    int n = blockIdx.x * 256 + threadIdx.x;          // over T*C
    int c = n & (Cc - 1);
    int t = n >> 8;
    int b = t >> 12;
    int hw = t & 4095;
    int h = hw >> 6, ww = hw & 63;
    const float* xb = x + ((size_t)b * Cc + c) * (Hh * Ww);
    float s = bias[c];
#pragma unroll
    for (int dh = -1; dh <= 1; ++dh)
#pragma unroll
        for (int dw = -1; dw <= 1; ++dw) {
            int h2 = h + dh, w2 = ww + dw;
            if (h2 >= 0 && h2 < Hh && w2 >= 0 && w2 < Ww)
                s += xb[h2 * Ww + w2] * w[c * 9 + (dh + 1) * 3 + (dw + 1)];
        }
    x1[(size_t)t * Cc + c] = xb[h * Ww + ww] + s;
}

// ---------------- LayerNorm over C, fp32 in -> bf16 out ----------------
__global__ __launch_bounds__(256) void ln_kernel(
        const float* __restrict__ x, const float* __restrict__ g,
        const float* __restrict__ b, unsigned short* __restrict__ out) {
    int t = blockIdx.x;
    int c = threadIdx.x;
    float v = x[(size_t)t * Cc + c];
    __shared__ float red[256];
    __shared__ float mu_s, rs_s;
    red[c] = v; __syncthreads();
    for (int s = 128; s > 0; s >>= 1) { if (c < s) red[c] += red[c + s]; __syncthreads(); }
    if (c == 0) mu_s = red[0] * (1.f / Cc);
    __syncthreads();
    float d = v - mu_s;
    red[c] = d * d; __syncthreads();
    for (int s = 128; s > 0; s >>= 1) { if (c < s) red[c] += red[c + s]; __syncthreads(); }
    if (c == 0) rs_s = rsqrtf(red[0] * (1.f / Cc) + 1e-6f);
    __syncthreads();
    out[(size_t)t * Cc + c] = f2bf(d * rs_s * g[c] + b[c]);
}

// ---------------- generic bf16 WMMA GEMM: Y[M,N] = A[M,K] * W[N,K]^T + bias ----------------
// Double-buffered async-to-LDS software pipeline: loads for tile k+1 overlap WMMAs on tile k.
// EPI: 0=qkv split(q bf16,k bf16,v f32) 1=+res->f32  2=gelu->bf16  3=+res->NCHW f32
template<int EPI>
__global__ __launch_bounds__(256) void gemm_bf16(
        const unsigned short* __restrict__ A, const unsigned short* __restrict__ W,
        const float* __restrict__ bias, const float* __restrict__ res,
        float* __restrict__ outF, unsigned short* __restrict__ outB,
        unsigned short* __restrict__ outB2, int N, int K) {
    int nBN = N >> 7;
    int bm = blockIdx.x / nBN, bn = blockIdx.x - bm * nBN;
    int m0 = bm << 7, n0 = bn << 7;
    __shared__ __align__(16) unsigned short As[2][128 * 32];
    __shared__ __align__(16) unsigned short Bs[2][128 * 32];
    int tid = threadIdx.x, lane = tid & 31, wave = tid >> 5;
    int wm = (wave >> 1) * 32;        // 4 wave-rows * 32 M
    int wn = (wave & 1) * 64;        // 2 wave-cols * 64 N
    unsigned asBase[2] = { lds_off(As[0]), lds_off(As[1]) };
    unsigned bsBase[2] = { lds_off(Bs[0]), lds_off(Bs[1]) };
    v8f acc[2][4];
    v8f zero = {0.f, 0.f, 0.f, 0.f, 0.f, 0.f, 0.f, 0.f};
#pragma unroll
    for (int i = 0; i < 2; ++i)
#pragma unroll
        for (int j = 0; j < 4; ++j) acc[i][j] = zero;

    // pipeline prologue: stage first tile
    stage_tiles(A, W, K, m0, n0, 0, asBase[0], bsBase[0], tid);
    wait_async0();
    __syncthreads();

    int cur = 0;
    for (int k0 = 0; k0 < K; k0 += 32) {
        // issue async loads for next tile into the other buffer (overlaps WMMAs)
        if (k0 + 32 < K)
            stage_tiles(A, W, K, m0, n0, k0 + 32, asBase[cur ^ 1], bsBase[cur ^ 1], tid);
        v16bf af0 = load_a_frag(As[cur], wm, lane);
        v16bf af1 = load_a_frag(As[cur], wm + 16, lane);
#pragma unroll
        for (int nt = 0; nt < 4; ++nt) {
            v16bf bf = load_b_frag(Bs[cur], wn + nt * 16, lane);
            acc[0][nt] = wmma_bf16(af0, bf, acc[0][nt]);
            acc[1][nt] = wmma_bf16(af1, bf, acc[1][nt]);
        }
        wait_async0();      // next tile landed (mostly hidden behind WMMAs)
        __syncthreads();    // all waves done reading cur + next tile visible
        cur ^= 1;
    }
    int col = lane & 15, hh = lane >> 4;
#pragma unroll
    for (int mt = 0; mt < 2; ++mt)
#pragma unroll
        for (int nt = 0; nt < 4; ++nt)
#pragma unroll
            for (int r8 = 0; r8 < 8; ++r8) {
                int gm = m0 + wm + mt * 16 + r8 + 8 * hh;
                int gn = n0 + wn + nt * 16 + col;
                float val = acc[mt][nt][r8] + bias[gn];
                if (EPI == 0) {
                    if (gn < 256)      outB [(size_t)gm * 256 + gn]       = f2bf(val);
                    else if (gn < 512) outB2[(size_t)gm * 256 + gn - 256] = f2bf(val);
                    else               outF [(size_t)gm * 256 + gn - 512] = val;
                } else if (EPI == 1) {
                    val += res[(size_t)gm * 256 + gn];
                    outF[(size_t)gm * 256 + gn] = val;
                } else if (EPI == 2) {
                    val = 0.5f * val * (1.0f + erff(val * 0.70710678118654752f));
                    outB[(size_t)gm * N + gn] = f2bf(val);
                } else { // EPI == 3: final residual, write NCHW output
                    val += res[(size_t)gm * 256 + gn];
                    int bI = gm >> 12;
                    int hw = gm & 4095;
                    outF[((size_t)bI * Cc + gn) * 4096 + hw] = val;
                }
            }
}

// ---------------- routing: per-region channel means ----------------
__global__ __launch_bounds__(256) void region_mean_kernel(
        const unsigned short* __restrict__ q, const unsigned short* __restrict__ k,
        float* __restrict__ qr, float* __restrict__ kr) {
    int br = blockIdx.x;                 // b*64 + r
    int b = br >> 6, r = br & 63;
    int c = threadIdx.x;
    int rh = r >> 3, rw = r & 7;
    float sq = 0.f, sk = 0.f;
    for (int i = 0; i < 64; ++i) {
        int h = rh * 8 + (i >> 3), w = rw * 8 + (i & 7);
        size_t t = ((size_t)b * Hh + h) * Ww + w;
        sq += bf2f(q[t * Cc + c]);
        sk += bf2f(k[t * Cc + c]);
    }
    qr[(size_t)br * Cc + c] = sq * (1.f / 64.f);
    kr[(size_t)br * Cc + c] = sk * (1.f / 64.f);
}

// ---------------- routing: region affinity + top-4 ----------------
__global__ __launch_bounds__(64) void route_topk_kernel(
        const float* __restrict__ qr, const float* __restrict__ kr, int* __restrict__ idx) {
    int br = blockIdx.x;
    int b = br >> 6;
    int j = threadIdx.x;
    __shared__ float a[64];
    const float* qp = qr + (size_t)br * Cc;
    const float* kp = kr + ((size_t)(b * 64 + j)) * Cc;
    float s = 0.f;
    for (int c = 0; c < Cc; ++c) s += qp[c] * kp[c];
    a[j] = s;
    __syncthreads();
    if (j == 0) {
        for (int t = 0; t < TOPK; ++t) {
            float best = -__builtin_inff(); int bi = 0;
            for (int jj = 0; jj < 64; ++jj) { float v = a[jj]; if (v > best) { best = v; bi = jj; } }
            idx[br * TOPK + t] = bi;
            a[bi] = -__builtin_inff();
        }
    }
}

// ---------------- gathered window attention (one block per (b, region)) ----------------
__global__ __launch_bounds__(256) void attn_kernel(
        const unsigned short* __restrict__ q, const unsigned short* __restrict__ k,
        const float* __restrict__ v, const int* __restrict__ idx, float* __restrict__ ao) {
    int b = blockIdx.x >> 6;
    int r = blockIdx.x & 63;
    int rh = r >> 3, rw = r & 7;
    __shared__ __align__(16) unsigned short Qs[RSQ * 32];        // 4 KB
    __shared__ __align__(16) unsigned short Ks[256 * 32];        // 16 KB (reused as Wsel/Isel)
    __shared__ __align__(16) float Vs[256 * 32];                 // 32 KB
    __shared__ __align__(16) float Ss[RSQ * 256];                // 64 KB
    float* Wsel = (float*)Ks;            // 64*32 floats (8 KB)
    int*   Isel = ((int*)Ks) + 2048;     // 64*32 ints   (8 KB)
    int tid = threadIdx.x, lane = tid & 31, wave = tid >> 5;
    const float scale = 0.0625f;         // 256^-0.5
    unsigned ksBase = lds_off(Ks);
    unsigned vsBase = lds_off(Vs);
    int idx4[TOPK];
#pragma unroll
    for (int t = 0; t < TOPK; ++t) idx4[t] = idx[(b * 64 + r) * TOPK + t];

    for (int m = 0; m < NHEAD; ++m) {
        // stage Q (scaled, bf16) - VALU path (needs scale)
        for (int e = tid; e < RSQ * 32; e += 256) {
            int i = e >> 5, d = e & 31;
            int h = rh * 8 + (i >> 3), w = rw * 8 + (i & 7);
            size_t t = ((size_t)b * Hh + h) * Ww + w;
            Qs[e] = f2bf(bf2f(q[t * Cc + m * DHEAD + d]) * scale);
        }
        // stage gathered K (bf16, b64/lane) and V (f32, b128/lane) via async-to-LDS
        for (int e8 = tid * 4; e8 < 256 * 32; e8 += 1024) {
            int j = e8 >> 5, d0 = e8 & 31;
            int rr = idx4[j >> 6], i = j & 63;
            int h = (rr >> 3) * 8 + (i >> 3), w = (rr & 7) * 8 + (i & 7);
            unsigned t = ((unsigned)b * Hh + h) * Ww + w;
            unsigned goff = t * Cc + (unsigned)(m * DHEAD + d0);
            async_ld_b64 (ksBase + (unsigned)e8 * 2u, goff * 2u, k);
            async_ld_b128(vsBase + (unsigned)e8 * 4u, goff * 4u, v);
        }
        wait_async0();
        __syncthreads();
        // scores: 4 M-tiles x 16 N-tiles, one 16x16x32 WMMA each
        int mt = wave & 3;          // M tile
        int nh = wave >> 2;         // N half: tiles [nh*8, nh*8+8)
        v16bf af = load_a_frag(Qs, mt * 16, lane);
        int col = lane & 15, hhalf = lane >> 4;
#pragma unroll
        for (int ntl = 0; ntl < 8; ++ntl) {
            int nt = nh * 8 + ntl;
            v16bf bfg = load_b_frag(Ks, nt * 16, lane);
            v8f az = {0.f, 0.f, 0.f, 0.f, 0.f, 0.f, 0.f, 0.f};
            az = wmma_bf16(af, bfg, az);
#pragma unroll
            for (int r8 = 0; r8 < 8; ++r8) {
                int row = mt * 16 + r8 + 8 * hhalf;
                Ss[row * 256 + nt * 16 + col] = az[r8];
            }
        }
        __syncthreads();
        // per query row: top-32 selection + softmax (threads 0..63)
        if (tid < 64) {
            float* row = Ss + tid * 256;
            for (int s = 0; s < KSEL; ++s) {
                float best = -__builtin_inff(); int bi = 0;
                for (int j = 0; j < 256; ++j) { float vv = row[j]; if (vv > best) { best = vv; bi = j; } }
                Wsel[tid * KSEL + s] = best;
                Isel[tid * KSEL + s] = bi;
                row[bi] = -__builtin_inff();
            }
            float mx = Wsel[tid * KSEL];
            float sum = 0.f;
            for (int s = 0; s < KSEL; ++s) {
                float e = __expf(Wsel[tid * KSEL + s] - mx);
                Wsel[tid * KSEL + s] = e;
                sum += e;
            }
            float inv = 1.f / sum;
            for (int s = 0; s < KSEL; ++s) Wsel[tid * KSEL + s] *= inv;
        }
        __syncthreads();
        // weighted sum of selected V rows
        for (int e = tid; e < RSQ * 32; e += 256) {
            int i = e >> 5, d = e & 31;
            float o = 0.f;
            for (int s = 0; s < KSEL; ++s)
                o += Wsel[i * KSEL + s] * Vs[Isel[i * KSEL + s] * 32 + d];
            int h = rh * 8 + (i >> 3), w = rw * 8 + (i & 7);
            size_t t = ((size_t)b * Hh + h) * Ww + w;
            ao[t * Cc + m * DHEAD + d] = o;
        }
        __syncthreads();
    }
}

// ---------------- LEPE: ao += dwconv5x5(v); emit bf16 for out-proj ----------------
__global__ __launch_bounds__(256) void lepe_kernel(
        const float* __restrict__ vf, const float* __restrict__ w,
        const float* __restrict__ bias, const float* __restrict__ ao,
        unsigned short* __restrict__ aob) {
    int n = blockIdx.x * 256 + threadIdx.x;   // over T*C
    int c = n & (Cc - 1);
    int t = n >> 8;
    int b = t >> 12;
    int hw = t & 4095;
    int h = hw >> 6, ww = hw & 63;
    float s = bias[c];
#pragma unroll
    for (int dh = -2; dh <= 2; ++dh)
#pragma unroll
        for (int dw = -2; dw <= 2; ++dw) {
            int h2 = h + dh, w2 = ww + dw;
            if (h2 >= 0 && h2 < Hh && w2 >= 0 && w2 < Ww)
                s += vf[(((size_t)b * Hh + h2) * Ww + w2) * Cc + c] * w[c * 25 + (dh + 2) * 5 + (dw + 2)];
        }
    aob[(size_t)t * Cc + c] = f2bf(ao[(size_t)t * Cc + c] + s);
}

// ---------------- host side ----------------
extern "C" void kernel_launch(void* const* d_in, const int* in_sizes, int n_in,
                              void* d_out, int out_size, void* d_ws, size_t ws_size,
                              hipStream_t stream) {
    const float* x      = (const float*)d_in[0];
    const float* pos_w  = (const float*)d_in[1];
    const float* pos_b  = (const float*)d_in[2];
    const float* ln1_g  = (const float*)d_in[3];
    const float* ln1_b  = (const float*)d_in[4];
    const float* qkv_w  = (const float*)d_in[5];
    const float* qkv_b  = (const float*)d_in[6];
    const float* lepe_w = (const float*)d_in[7];
    const float* lepe_b = (const float*)d_in[8];
    const float* out_w  = (const float*)d_in[9];
    const float* out_b  = (const float*)d_in[10];
    const float* ln2_g  = (const float*)d_in[11];
    const float* ln2_b  = (const float*)d_in[12];
    const float* mlp_w1 = (const float*)d_in[13];
    const float* mlp_b1 = (const float*)d_in[14];
    const float* mlp_w2 = (const float*)d_in[15];
    const float* mlp_b2 = (const float*)d_in[16];
    float* out = (float*)d_out;

    char* ws = (char*)d_ws;
    const size_t TC4 = (size_t)Tt * Cc * 4;          // 33.5 MB
    const size_t TC2 = (size_t)Tt * Cc * 2;          // 16.8 MB
    float*          x1   = (float*)(ws + 0);                     // residual stream (NHWC)
    unsigned short* abf  = (unsigned short*)(ws + TC4);          // LN1 out bf16
    unsigned short* qbf  = (unsigned short*)(ws + TC4 + TC2);
    unsigned short* kbf  = (unsigned short*)(ws + TC4 + 2 * TC2);
    float*          vf   = (float*)(ws + TC4 + 3 * TC2);
    float*          ao   = (float*)(ws + 2 * TC4 + 3 * TC2);
    unsigned short* aobf = (unsigned short*)(ws + 3 * TC4 + 3 * TC2);
    float*          x2   = (float*)(ws + 3 * TC4 + 4 * TC2);
    unsigned short* b2bf = (unsigned short*)(ws + 4 * TC4 + 4 * TC2);
    unsigned short* hbf  = abf;   // overlay: abf/qbf/kbf dead by MLP time (exactly 50.3 MB)
    char* small = ws + 4 * TC4 + 5 * TC2;
    float*          qr   = (float*)(small);                       // [B*64, 256]
    float*          kr   = (float*)(small + 524288);
    int*            ridx = (int*)(small + 1048576);               // [B*64, 4]
    unsigned short* wqkv = (unsigned short*)(small + 1048576 + 8192);
    unsigned short* wout = (unsigned short*)((char*)wqkv + 768 * 256 * 2);
    unsigned short* w1t  = (unsigned short*)((char*)wout + 256 * 256 * 2);
    unsigned short* w2t  = (unsigned short*)((char*)w1t + 768 * 256 * 2);

    // weights -> bf16 (transpose MLP weights to [N,K])
    cvt_bf16<<<768, 256, 0, stream>>>(qkv_w, wqkv, 768 * 256);
    cvt_bf16<<<256, 256, 0, stream>>>(out_w, wout, 256 * 256);
    cvt_bf16_T<<<768, 256, 0, stream>>>(mlp_w1, w1t, 256, 768);
    cvt_bf16_T<<<768, 256, 0, stream>>>(mlp_w2, w2t, 768, 256);

    // x1 = x + posconv(x)   (NHWC)
    posconv_kernel<<<(Tt * Cc) / 256, 256, 0, stream>>>(x, pos_w, pos_b, x1);
    // LN1 -> bf16
    ln_kernel<<<Tt, 256, 0, stream>>>(x1, ln1_g, ln1_b, abf);
    // QKV GEMM: [T,768] = abf * wqkv^T
    gemm_bf16<0><<<(Tt / 128) * (768 / 128), 256, 0, stream>>>(
        abf, wqkv, qkv_b, nullptr, vf, qbf, kbf, 768, 256);
    // routing
    region_mean_kernel<<<Bn * NREG, 256, 0, stream>>>(qbf, kbf, qr, kr);
    route_topk_kernel<<<Bn * NREG, 64, 0, stream>>>(qr, kr, ridx);
    // attention
    attn_kernel<<<Bn * NREG, 256, 0, stream>>>(qbf, kbf, vf, ridx, ao);
    // lepe + bf16 convert
    lepe_kernel<<<(Tt * Cc) / 256, 256, 0, stream>>>(vf, lepe_w, lepe_b, ao, aobf);
    // out projection + residual -> x2
    gemm_bf16<1><<<(Tt / 128) * (256 / 128), 256, 0, stream>>>(
        aobf, wout, out_b, x1, x2, nullptr, nullptr, 256, 256);
    // LN2 -> bf16
    ln_kernel<<<Tt, 256, 0, stream>>>(x2, ln2_g, ln2_b, b2bf);
    // MLP1 (GELU) -> hbf
    gemm_bf16<2><<<(Tt / 128) * (768 / 128), 256, 0, stream>>>(
        b2bf, w1t, mlp_b1, nullptr, nullptr, hbf, nullptr, 768, 256);
    // MLP2 + residual -> d_out (NCHW)
    gemm_bf16<3><<<(Tt / 128) * (256 / 128), 256, 0, stream>>>(
        hbf, w2t, mlp_b2, x2, out, nullptr, nullptr, 256, 768);
}
